// CompressiveMemory_76373108457893
// MI455X (gfx1250) — matrix operand order repaired and checked by reading, not compiled, not run on hardware
//
#include <hip/hip_runtime.h>

#define B_    8
#define S_    8192
#define D_    512
#define H_    8
#define DK_   64
#define DV_   64
#define L_    512
#define NSEG_ 16

typedef __attribute__((ext_vector_type(16))) __bf16 bf16x16;
typedef __attribute__((ext_vector_type(8)))  float  f32x8;
typedef __attribute__((ext_vector_type(4)))  int    v4i;

// ---- CDNA5 async global->LDS copy (guarded; falls back to sync copy) -------
#if defined(__gfx1250__) && __has_builtin(__builtin_amdgcn_global_load_async_to_lds_b128)
#define HAVE_ASYNC_LDS 1
#else
#define HAVE_ASYNC_LDS 0
#endif

static __device__ __forceinline__ void copy16_to_lds(void* lds, const void* g) {
#if HAVE_ASYNC_LDS
  typedef __attribute__((address_space(1))) v4i gv4i;   // global int4
  typedef __attribute__((address_space(3))) v4i lv4i;   // LDS int4
  // generic->AS via integer: LDS generic addr low 32 bits are the LDS offset
  __builtin_amdgcn_global_load_async_to_lds_b128(
      (gv4i*)(unsigned long long)(const char*)g,
      (lv4i*)(unsigned int)(unsigned long long)(char*)lds, 0, 0);
#else
  *reinterpret_cast<uint4*>(lds) = *reinterpret_cast<const uint4*>(g);
#endif
}
static __device__ __forceinline__ void async_copy_wait() {
#if HAVE_ASYNC_LDS
#if __has_builtin(__builtin_amdgcn_s_wait_asynccnt)
  __builtin_amdgcn_s_wait_asynccnt(0);
#else
  asm volatile("s_wait_asynccnt 0x0" ::: "memory");
#endif
#endif
}

static __device__ __forceinline__ float elu1(float x) {
  return x > 0.f ? x + 1.f : __expf(x);
}

// A fragment (16x32 bf16): lane r=lane&15 holds row r; half h=lane>>4 holds
// K chunks [8h..8h+7] and [16+8h..16+8h+7]  (two b128 LDS reads).
static __device__ __forceinline__ bf16x16 ld_frag_a(const __bf16* rowk0, int half) {
  bf16x16 f;
  uint4* p = reinterpret_cast<uint4*>(&f);
  p[0] = *reinterpret_cast<const uint4*>(rowk0 + 8 * half);
  p[1] = *reinterpret_cast<const uint4*>(rowk0 + 16 + 8 * half);
  return f;
}
// B fragment (32x16 bf16), column-major staged: lane n=lane&15 holds col n;
// half h holds K chunks [8h..8h+7] and [16+8h..16+8h+7] (two b128 LDS reads).
static __device__ __forceinline__ bf16x16 ld_frag_b(const __bf16* colk0, int half) {
  bf16x16 f;
  uint4* p = reinterpret_cast<uint4*>(&f);
  p[0] = *reinterpret_cast<const uint4*>(colk0 + 8 * half);
  p[1] = *reinterpret_cast<const uint4*>(colk0 + 16 + 8 * half);
  return f;
}
// sigma(q)=elu(q)+1 applied element-wise while building the A fragment
static __device__ __forceinline__ bf16x16 ld_frag_a_elu(const __bf16* rowk0, int half) {
  bf16x16 f;
  const __bf16* p0 = rowk0 + 8 * half;
  const __bf16* p1 = rowk0 + 16 + 8 * half;
#pragma unroll
  for (int i = 0; i < 8; ++i) { float x = (float)p0[i]; f[i]     = (__bf16)elu1(x); }
#pragma unroll
  for (int i = 0; i < 8; ++i) { float x = (float)p1[i]; f[8 + i] = (__bf16)elu1(x); }
  return f;
}
static __device__ __forceinline__ f32x8 wmma_bf16(bf16x16 a, bf16x16 b, f32x8 c) {
  return __builtin_amdgcn_wmma_f32_16x16x32_bf16(false, a, false, b, (short)0, c, false, false);
}

// ---------------------------------------------------------------------------
// K1: fused Q/K/V projection GEMM  (65536x512) @ (512x512), bf16 WMMA
// grid (512, 4, 3), block 256 (8 waves, 2x4 wave grid, 64x32 per wave)
// K-step 64 per staging round (2 WMMA k-steps per barrier pair)
// ---------------------------------------------------------------------------
__global__ __launch_bounds__(256) void k_proj(
    const float* __restrict__ x, const float* __restrict__ Wq,
    const float* __restrict__ Wk, const float* __restrict__ Wv,
    __bf16* __restrict__ Qb, __bf16* __restrict__ Kb, __bf16* __restrict__ Vb) {
  __shared__ __align__(16) __bf16 As[128 * 72];   // [row][k] stride 72
  __shared__ __align__(16) __bf16 Bt[128 * 72];   // [n][k]   stride 72
  const int t = threadIdx.x;
  const int lane = t & 31, wv = t >> 5;
  const int half = lane >> 4, lr = lane & 15;
  const int m0 = blockIdx.x * 128;
  const int n0 = blockIdx.y * 128;
  const int which = blockIdx.z;
  const float* W = which == 0 ? Wq : (which == 1 ? Wk : Wv);
  __bf16* outp = which == 0 ? Qb : (which == 1 ? Kb : Vb);
  const int wm = wv >> 2, wn = wv & 3;

  f32x8 acc[4][2] = {};

  for (int k0 = 0; k0 < D_; k0 += 64) {
#pragma unroll
    for (int it = 0; it < 8; ++it) {             // A: 128x64 f32 -> bf16
      int idx = t + it * 256;                    // 2048 float4s
      int row = idx >> 4, c4 = (idx & 15) * 4;
      const float4 v = *reinterpret_cast<const float4*>(
          &x[(size_t)(m0 + row) * D_ + k0 + c4]);
      __bf16* dst = &As[row * 72 + c4];
      dst[0] = (__bf16)v.x; dst[1] = (__bf16)v.y;
      dst[2] = (__bf16)v.z; dst[3] = (__bf16)v.w;
    }
#pragma unroll
    for (int it = 0; it < 8; ++it) {             // B: 64x128 f32 -> bf16, transposed
      int idx = t + it * 256;                    // 2048 float4s
      int kk = idx >> 5, n4 = (idx & 31) * 4;
      const float4 v = *reinterpret_cast<const float4*>(
          &W[(size_t)(k0 + kk) * D_ + n0 + n4]);
      Bt[(n4 + 0) * 72 + kk] = (__bf16)v.x;
      Bt[(n4 + 1) * 72 + kk] = (__bf16)v.y;
      Bt[(n4 + 2) * 72 + kk] = (__bf16)v.z;
      Bt[(n4 + 3) * 72 + kk] = (__bf16)v.w;
    }
    __syncthreads();
#pragma unroll
    for (int ks = 0; ks < 2; ++ks) {
      bf16x16 afr[4], bfr[2];
#pragma unroll
      for (int mt = 0; mt < 4; ++mt)
        afr[mt] = ld_frag_a(&As[(wm * 64 + mt * 16 + lr) * 72 + ks * 32], half);
#pragma unroll
      for (int nt = 0; nt < 2; ++nt)
        bfr[nt] = ld_frag_b(&Bt[(wn * 32 + nt * 16 + lr) * 72 + ks * 32], half);
#pragma unroll
      for (int mt = 0; mt < 4; ++mt)
#pragma unroll
        for (int nt = 0; nt < 2; ++nt)
          acc[mt][nt] = wmma_bf16(afr[mt], bfr[nt], acc[mt][nt]);
    }
    __syncthreads();
  }
#pragma unroll
  for (int mt = 0; mt < 4; ++mt)
#pragma unroll
    for (int nt = 0; nt < 2; ++nt) {
      int col = n0 + wn * 32 + nt * 16 + lr;
#pragma unroll
      for (int e = 0; e < 8; ++e) {
        int row = m0 + wm * 64 + mt * 16 + e + 8 * half;
        outp[(size_t)row * D_ + col] = (__bf16)acc[mt][nt][e];
      }
    }
}

// ---------------------------------------------------------------------------
// K2: per-(b,h,seg) partials: Sk = sigma(K)^T @ V (64x64), zs = sum_l sigma(K)
// grid 1024 (bh*16+seg), block 256. Output: [bx]*4160 floats (4096 mem + 64 z)
// ---------------------------------------------------------------------------
__global__ __launch_bounds__(256) void k_segsum(
    const __bf16* __restrict__ Kb, const __bf16* __restrict__ Vb,
    float* __restrict__ Sk) {
  __shared__ __align__(16) __bf16 SkT[64 * 72];   // [dk][l] stride 72
  __shared__ __align__(16) __bf16 VtT[64 * 72];   // [dv][l] stride 72
  __shared__ float zacc[64];
  const int t = threadIdx.x;
  const int lane = t & 31, wv = t >> 5;
  const int half = lane >> 4, lr = lane & 15;
  const int bx = blockIdx.x;
  const int bh = bx >> 4, seg = bx & 15;
  const int b = bh >> 3, hd = bh & 7;
  const size_t base = ((size_t)(b * S_ + hd * (S_ / H_))) * D_ + (size_t)seg * (L_ * DK_);
  const __bf16* Kg = Kb + base;
  const __bf16* Vg = Vb + base;
  if (t < 64) zacc[t] = 0.f;

  f32x8 acc[2] = {};
  for (int l0 = 0; l0 < L_; l0 += 64) {
    __syncthreads();
#pragma unroll
    for (int it = 0; it < 16; ++it) {
      int e = t + it * 256;                   // 0..4095
      int li = e >> 6, kk = e & 63;
      float kv = (float)Kg[(size_t)(l0 + li) * DK_ + kk];
      float sg = elu1(kv);
      SkT[kk * 72 + li] = (__bf16)sg;
      atomicAdd(&zacc[kk], sg);
      VtT[kk * 72 + li] = Vg[(size_t)(l0 + li) * DV_ + kk];
    }
    __syncthreads();
#pragma unroll
    for (int ti = 0; ti < 2; ++ti) {
      int tl = 2 * wv + ti;
      int mt = tl >> 2, nt = tl & 3;
#pragma unroll
      for (int ks = 0; ks < 2; ++ks) {
        bf16x16 a  = ld_frag_a(&SkT[(mt * 16 + lr) * 72 + ks * 32], half);
        bf16x16 bb = ld_frag_b(&VtT[(nt * 16 + lr) * 72 + ks * 32], half);
        acc[ti] = wmma_bf16(a, bb, acc[ti]);
      }
    }
  }
  __syncthreads();
  float* out = Sk + (size_t)bx * 4160;
#pragma unroll
  for (int ti = 0; ti < 2; ++ti) {
    int tl = 2 * wv + ti;
    int mt = tl >> 2, nt = tl & 3;
    int col = nt * 16 + lr;
#pragma unroll
    for (int e = 0; e < 8; ++e) {
      int row = mt * 16 + e + 8 * half;
      out[row * 64 + col] = acc[ti][e];
    }
  }
  if (t < 64) out[4096 + t] = zacc[t];
}

// ---------------------------------------------------------------------------
// K3: exclusive prefix over 16 segments per (b,h), in place. mem0=0, z0=1/64
// ---------------------------------------------------------------------------
__global__ __launch_bounds__(256) void k_scan(float* __restrict__ Sk) {
  const int bh = blockIdx.x;
  for (int e = threadIdx.x; e < 4160; e += 256) {
    float run = (e < 4096) ? 0.f : (1.f / 64.f);
    for (int s = 0; s < NSEG_; ++s) {
      float* p = Sk + ((size_t)(bh * NSEG_ + s)) * 4160 + e;
      float c = *p;
      *p = run;
      run += c;
    }
  }
}

// ---------------------------------------------------------------------------
// K4: per-(b,h,seg) attention: causal softmax(QK^T/8)V + sigma(Q)Mem/(sigma(Q)z)
// grid 1024, block 256 (8 waves); 16 row-blocks of 32 query rows
// ---------------------------------------------------------------------------
__global__ __launch_bounds__(256) void k_attn(
    const __bf16* __restrict__ Qb, const __bf16* __restrict__ Kb,
    const __bf16* __restrict__ Vb, const float* __restrict__ Skp,
    const float* __restrict__ betas, __bf16* __restrict__ attb) {
  __shared__ __align__(16) __bf16 Ks[512 * 72];    // [j][dk]   stride 72
  __shared__ __align__(16) __bf16 Vt[64 * 520];    // [dv][j]   stride 520
  __shared__ __align__(16) __bf16 Sc[32 * 520];    // scores/P  stride 520
  __shared__ __align__(16) __bf16 Qs[32 * 72];     // [row][dk] stride 72
  __shared__ __align__(16) __bf16 MemT[64 * 72];   // [dv][dk]  stride 72
  __shared__ float zsh[64], betash[64];
  __shared__ float red[32 * 8], red2[32 * 8];
  __shared__ float rowmax[32], rowsuminv[32], deninv[32];

  const int t = threadIdx.x;
  const int lane = t & 31, wv = t >> 5;
  const int half = lane >> 4, lr = lane & 15;
  const int bx = blockIdx.x;
  const int bh = bx >> 4, seg = bx & 15;
  const int b = bh >> 3, hd = bh & 7;
  const size_t base = ((size_t)(b * S_ + hd * (S_ / H_))) * D_ + (size_t)seg * (L_ * DK_);
  const __bf16* Qg = Qb + base;
  const __bf16* Kg = Kb + base;
  const __bf16* Vg = Vb + base;
  const float* memp = Skp + (size_t)bx * 4160;
  __bf16* attG = attb + ((size_t)(seg * B_ + b)) * (L_ * H_ * DV_) + (size_t)hd * (L_ * DV_);

  // ---- stage K via async-to-LDS, overlap with V transpose + Mem^T ----------
#pragma unroll
  for (int it = 0; it < 16; ++it) {
    int idx = t + it * 256;                       // 8-elt chunks: 512*64/8
    int j = idx >> 3, c8 = (idx & 7) * 8;
    copy16_to_lds(&Ks[j * 72 + c8], &Kg[(size_t)j * DK_ + c8]);
  }
#pragma unroll
  for (int it = 0; it < 16; ++it) {               // V transposed (VALU work)
    int idx = t + it * 256;
    int j = idx >> 3, c8 = (idx & 7) * 8;
    const __bf16* src = &Vg[(size_t)j * DV_ + c8];
#pragma unroll
    for (int u = 0; u < 8; ++u) Vt[(c8 + u) * 520 + j] = src[u];
  }
#pragma unroll
  for (int it = 0; it < 16; ++it) {               // Mem^T f32->bf16
    int e = t + it * 256;                         // 0..4095
    int kk = e >> 6, vv = e & 63;
    MemT[vv * 72 + kk] = (__bf16)memp[e];
  }
  if (t < 64) {
    zsh[t] = memp[4096 + t];
    betash[t] = 1.f / (1.f + __expf(-betas[hd * 64 + t]));
  }
  async_copy_wait();

  for (int rb = 0; rb < 16; ++rb) {
    const int r0 = rb * 32;
    __syncthreads();
    {                                             // stage 32 query rows (async)
      int row = t >> 3, c8 = (t & 7) * 8;
      copy16_to_lds(&Qs[row * 72 + c8], &Qg[(size_t)(r0 + row) * DK_ + c8]);
      async_copy_wait();
    }
    __syncthreads();
    {                                             // scores (32 x 512), masked
      int mt = wv & 1;
#pragma unroll
      for (int q = 0; q < 8; ++q) {
        int nt = (wv >> 1) + q * 4;
        f32x8 acc = {};
#pragma unroll
        for (int ks = 0; ks < 2; ++ks) {
          bf16x16 a  = ld_frag_a(&Qs[(mt * 16 + lr) * 72 + ks * 32], half);
          bf16x16 bb = ld_frag_b(&Ks[(nt * 16 + lr) * 72 + ks * 32], half);
          acc = wmma_bf16(a, bb, acc);
        }
        int j = nt * 16 + lr;
#pragma unroll
        for (int e = 0; e < 8; ++e) {
          int m = mt * 16 + e + 8 * half;
          float s = acc[e] * 0.125f;
          if (j > r0 + m) s = -1e30f;             // causal within segment
          Sc[m * 520 + j] = (__bf16)s;
        }
      }
    }
    __syncthreads();
    {                                             // softmax stats + denominators
      int row = t >> 3, sub = t & 7;
      float mx = -3.0e38f;
      for (int jj = 0; jj < 64; ++jj)
        mx = fmaxf(mx, (float)Sc[row * 520 + sub * 64 + jj]);
      red[row * 8 + sub] = mx;
      __syncthreads();
      if (sub == 0) {
        float m2 = red[row * 8];
        for (int u = 1; u < 8; ++u) m2 = fmaxf(m2, red[row * 8 + u]);
        rowmax[row] = m2;
      }
      __syncthreads();
      float m2 = rowmax[row], sum = 0.f;
      for (int jj = 0; jj < 64; ++jj) {
        int jc = sub * 64 + jj;
        float e = __expf((float)Sc[row * 520 + jc] - m2);
        Sc[row * 520 + jc] = (__bf16)e;           // unnormalized P
        sum += e;
      }
      red[row * 8 + sub] = sum;
      float dpart = 0.f;                          // sigma(q) . z partial
      for (int u = 0; u < 8; ++u) {
        int kk = sub * 8 + u;
        dpart += elu1((float)Qs[row * 72 + kk]) * zsh[kk];
      }
      red2[row * 8 + sub] = dpart;
      __syncthreads();
      if (sub == 0) {
        float s2 = 0.f, d2 = 0.f;
        for (int u = 0; u < 8; ++u) { s2 += red[row * 8 + u]; d2 += red2[row * 8 + u]; }
        rowsuminv[row] = 1.f / s2;
        deninv[row]    = 1.f / d2;
      }
    }
    __syncthreads();
    {                                             // P@V and sigma(Q)@Mem, combine
      int mt = wv & 1, nt = wv >> 1;
      f32x8 dacc = {};
#pragma unroll
      for (int ks = 0; ks < 16; ++ks) {
        bf16x16 a  = ld_frag_a(&Sc[(mt * 16 + lr) * 520 + ks * 32], half);
        bf16x16 bb = ld_frag_b(&Vt[(nt * 16 + lr) * 520 + ks * 32], half);
        dacc = wmma_bf16(a, bb, dacc);
      }
      f32x8 macc = {};
#pragma unroll
      for (int ks = 0; ks < 2; ++ks) {
        bf16x16 a  = ld_frag_a_elu(&Qs[(mt * 16 + lr) * 72 + ks * 32], half);
        bf16x16 bb = ld_frag_b(&MemT[(nt * 16 + lr) * 72 + ks * 32], half);
        macc = wmma_bf16(a, bb, macc);
      }
      int vcol = nt * 16 + lr;
      float bta = betash[vcol];
#pragma unroll
      for (int e = 0; e < 8; ++e) {
        int m = mt * 16 + e + 8 * half;
        float dot = dacc[e] * rowsuminv[m];
        float mem = macc[e] * deninv[m];
        attG[(size_t)(r0 + m) * DV_ + vcol] = (__bf16)(bta * mem + (1.f - bta) * dot);
      }
    }
  }
}

// ---------------------------------------------------------------------------
// K5: output projection (65536x512) @ Wout(512x512) -> f32, bf16 WMMA
// grid (512, 4), block 256; K-step 64, A tile staged via async-to-LDS
// ---------------------------------------------------------------------------
__global__ __launch_bounds__(256) void k_outproj(
    const __bf16* __restrict__ attb, const float* __restrict__ Wout,
    float* __restrict__ out) {
  __shared__ __align__(16) __bf16 As[128 * 72];
  __shared__ __align__(16) __bf16 Bt[128 * 72];
  const int t = threadIdx.x;
  const int lane = t & 31, wv = t >> 5;
  const int half = lane >> 4, lr = lane & 15;
  const int m0 = blockIdx.x * 128;
  const int n0 = blockIdx.y * 128;
  const int wm = wv >> 2, wn = wv & 3;
  // rows m0..m0+127 share (b, seg): m = b*8192 + seg*512 + l
  const int bB  = m0 >> 13;
  const int s0  = m0 & 8191;
  const int seg = s0 >> 9;
  const int l0  = s0 & 511;
  const __bf16* Arows = attb + ((size_t)(seg * B_ + bB)) * (512 * 512) + (size_t)l0 * 512;

  f32x8 acc[4][2] = {};
  for (int k0 = 0; k0 < 512; k0 += 64) {
#pragma unroll
    for (int it = 0; it < 4; ++it) {              // A already bf16 (async copy)
      int idx = t + it * 256;                     // 1024 chunks of 8 (128*64/8)
      int row = idx >> 3, c8 = (idx & 7) * 8;
      copy16_to_lds(&As[row * 72 + c8], &Arows[(size_t)row * 512 + k0 + c8]);
    }
#pragma unroll
    for (int it = 0; it < 8; ++it) {              // B from Wout f32, transposed
      int idx = t + it * 256;
      int kk = idx >> 5, n4 = (idx & 31) * 4;
      const float4 v = *reinterpret_cast<const float4*>(
          &Wout[(size_t)(k0 + kk) * 512 + n0 + n4]);
      Bt[(n4 + 0) * 72 + kk] = (__bf16)v.x;
      Bt[(n4 + 1) * 72 + kk] = (__bf16)v.y;
      Bt[(n4 + 2) * 72 + kk] = (__bf16)v.z;
      Bt[(n4 + 3) * 72 + kk] = (__bf16)v.w;
    }
    async_copy_wait();
    __syncthreads();
#pragma unroll
    for (int ks = 0; ks < 2; ++ks) {
      bf16x16 afr[4], bfr[2];
#pragma unroll
      for (int mt = 0; mt < 4; ++mt)
        afr[mt] = ld_frag_a(&As[(wm * 64 + mt * 16 + lr) * 72 + ks * 32], half);
#pragma unroll
      for (int nt = 0; nt < 2; ++nt)
        bfr[nt] = ld_frag_b(&Bt[(wn * 32 + nt * 16 + lr) * 72 + ks * 32], half);
#pragma unroll
      for (int mt = 0; mt < 4; ++mt)
#pragma unroll
        for (int nt = 0; nt < 2; ++nt)
          acc[mt][nt] = wmma_bf16(afr[mt], bfr[nt], acc[mt][nt]);
    }
    __syncthreads();
  }
#pragma unroll
  for (int mt = 0; mt < 4; ++mt)
#pragma unroll
    for (int nt = 0; nt < 2; ++nt) {
      int col = n0 + wn * 32 + nt * 16 + lr;
#pragma unroll
      for (int e = 0; e < 8; ++e) {
        int row = m0 + wm * 64 + mt * 16 + e + 8 * half;
        out[(size_t)row * 512 + col] = acc[mt][nt][e];
      }
    }
}

// ---------------------------------------------------------------------------
extern "C" void kernel_launch(void* const* d_in, const int* in_sizes, int n_in,
                              void* d_out, int out_size, void* d_ws, size_t ws_size,
                              hipStream_t stream) {
  (void)in_sizes; (void)n_in; (void)out_size; (void)ws_size;
  const float* x     = (const float*)d_in[0];
  const float* Wq    = (const float*)d_in[1];
  const float* Wk    = (const float*)d_in[2];
  const float* Wv    = (const float*)d_in[3];
  const float* Wout  = (const float*)d_in[4];
  const float* betas = (const float*)d_in[5];

  char* ws = (char*)d_ws;
  const size_t szProj = (size_t)B_ * S_ * 512 * sizeof(__bf16);  // 64 MiB each
  __bf16* Qb   = (__bf16*)(ws + 0 * szProj);
  __bf16* Kb   = (__bf16*)(ws + 1 * szProj);
  __bf16* Vb   = (__bf16*)(ws + 2 * szProj);
  __bf16* attb = (__bf16*)(ws + 3 * szProj);
  float*  Sk   = (float*) (ws + 4 * szProj);    // 1024 * 4160 f32

  k_proj   <<<dim3(512, 4, 3), dim3(256), 0, stream>>>(x, Wq, Wk, Wv, Qb, Kb, Vb);
  k_segsum <<<dim3(1024),      dim3(256), 0, stream>>>(Kb, Vb, Sk);
  k_scan   <<<dim3(64),        dim3(256), 0, stream>>>(Sk);
  k_attn   <<<dim3(1024),      dim3(256), 0, stream>>>(Qb, Kb, Vb, Sk, betas, attb);
  k_outproj<<<dim3(512, 4),    dim3(256), 0, stream>>>(attb, Wout, (float*)d_out);
}
